// GemNetWrapper_27041114095751
// MI455X (gfx1250) — compile-verified, matrix operand order checked by reading, not run on hardware
//
#include <hip/hip_runtime.h>
#include <hip/hip_bf16.h>

// ---------------------------------------------------------------------------
// GemNet-style GNN forward, CDNA5 (gfx1250) WMMA implementation, v2.
// All GEMMs run as v_wmma_f32_16x16x32_f16 (f16 in, f32 accumulate).
// Each wave owns TWO 16-row M-subtiles so every B (weight) fragment loaded
// from L2 feeds two WMMAs -> halves weight traffic, doubles WMMA density.
// ---------------------------------------------------------------------------

typedef __attribute__((ext_vector_type(16))) _Float16 v16h;
typedef __attribute__((ext_vector_type(8)))  _Float16 v8h;
typedef __attribute__((ext_vector_type(8)))  float    v8f;

#define N_NODES  20000
#define N_EDGES  160000
#define N_GRAPHS 64
#define HID      128
#define RADIAL   50
#define NLAYERS  3
#define CUTOFF   10.0f
#define KP1      320   // 2*HID + RADIAL (306) padded to multiple of 32
#define KU       256   // 2*HID

__device__ __forceinline__ float silu_f(float x) { return x / (1.0f + __expf(-x)); }

__device__ __forceinline__ v8f wmma16(v16h a, v16h b, v8f c) {
  return __builtin_amdgcn_wmma_f32_16x16x32_f16(false, a, false, b, (short)0, c,
                                                false, false);
}

// Load one lane's 16-half fragment in the CDNA5 16-bit A/B layout.
// Lane L owns row L%16; K-chunks: base=(L/16)*8, halves [base..base+7] and
// [base+16..base+23] within each 32-wide K step. Both chunks are 16B aligned.
__device__ __forceinline__ v16h frag_ld(const _Float16* rowp, int ks, int lane) {
  int kb = ks * 32 + ((lane >> 4) << 3);
  v8h lo = *(const v8h*)(rowp + kb);
  v8h hi = *(const v8h*)(rowp + kb + 16);
  v16h r;
#pragma unroll
  for (int i = 0; i < 8; ++i) { r[i] = lo[i]; r[i + 8] = hi[i]; }
  return r;
}

// ---------------------------------------------------------------------------
// Weight prep: f32 [K,128] (per-mat) -> f16 transposed [128][K]
// ---------------------------------------------------------------------------
__global__ void prep_w1(const float* __restrict__ W, _Float16* __restrict__ Wt) {
  int i = blockIdx.x * blockDim.x + threadIdx.x;
  if (i >= NLAYERS * HID * KP1) return;
  int k = i % KP1, n = (i / KP1) % HID, l = i / (KP1 * HID);
  float v = (k < 2 * HID + RADIAL) ? W[((size_t)l * (2 * HID + RADIAL) + k) * HID + n] : 0.f;
  Wt[i] = (_Float16)v;
}

__global__ void prep_t(const float* __restrict__ W, _Float16* __restrict__ Wt,
                       int Kin, int total) {
  int i = blockIdx.x * blockDim.x + threadIdx.x;
  if (i >= total) return;
  int k = i % Kin, n = (i / Kin) % HID, m = i / (Kin * HID);
  Wt[i] = (_Float16)W[((size_t)m * Kin + k) * HID + n];
}

__global__ void embed_kernel(const int* __restrict__ an, const float* __restrict__ emb,
                             float* __restrict__ x, _Float16* __restrict__ xh) {
  size_t i = (size_t)blockIdx.x * blockDim.x + threadIdx.x;
  if (i >= (size_t)N_NODES * HID) return;
  int n = (int)(i / HID), h = (int)(i % HID);
  float v = emb[(size_t)an[n] * HID + h];
  x[i] = v; xh[i] = (_Float16)v;
}

__global__ void zero_f(float* __restrict__ p, size_t n) {
  size_t i = (size_t)blockIdx.x * blockDim.x + threadIdx.x;
  if (i < n) p[i] = 0.f;
}

// ---------------------------------------------------------------------------
// Edge kernel: one wave owns a 32-edge tile (2 x 16-row WMMA subtiles).
//   ef[32x320] -> silu -> h1[32x128] -> silu -> h2[32x128] -> msg -> atomic agg
// ---------------------------------------------------------------------------
__global__ __launch_bounds__(64) void edge_kernel(
    const float* __restrict__ dist,
    const int* __restrict__ erow, const int* __restrict__ ecol,
    const _Float16* __restrict__ xh,
    const _Float16* __restrict__ Wt1, const float* __restrict__ b1,
    const _Float16* __restrict__ Wt2, const float* __restrict__ b2,
    const _Float16* __restrict__ Wt3, const float* __restrict__ b3,
    float* __restrict__ agg) {
  __shared__ _Float16 s_ef[2][32][KP1];   // 40 KB
  __shared__ _Float16 s_h [2][32][HID];   // 16 KB
  __shared__ int      s_col[2][32];

  const int lane = threadIdx.x & 31;
  const int wid  = threadIdx.x >> 5;
  _Float16 (*ef)[KP1] = s_ef[wid];
  _Float16 (*hs)[HID] = s_h[wid];
  _Float16* h2 = &s_ef[wid][0][0];        // ef region reused after stage 1 [32][HID]
  int* cols = s_col[wid];

  const int ntile = N_EDGES / 32;
  const int gw0 = blockIdx.x * 2 + wid;
  const int gws = gridDim.x * 2;
  const float delta = CUTOFF / (RADIAL - 1);
  const float coeff = -0.5f / (delta * delta);
  const int m16 = lane & 15;
  const int mhi = (lane >> 4) << 3;

  for (int t = gw0; t < ntile; t += gws) {
    const int e0 = t * 32;
    // ---- stage 0: gather [x[row] | x[col] | rbf(d) | pad] into LDS --------
    for (int e = 0; e < 32; ++e) {
      int r = erow[e0 + e], c = ecol[e0 + e];
      if (lane == 0) cols[e] = c;
      const unsigned* sr = (const unsigned*)(xh + (size_t)r * HID);
      const unsigned* sc = (const unsigned*)(xh + (size_t)c * HID);
      unsigned* dr = (unsigned*)&ef[e][0];
      dr[lane]      = sr[lane];
      dr[lane + 32] = sr[lane + 32];
      dr[lane + 64] = sc[lane];
      dr[lane + 96] = sc[lane + 32];
      float d = dist[e0 + e];
#pragma unroll
      for (int j = lane; j < 64; j += 32) {
        float dv = d - (float)j * delta;
        float v = (j < RADIAL) ? __expf(coeff * dv * dv) : 0.f;
        ef[e][2 * HID + j] = (_Float16)v;
      }
    }
    asm volatile("s_wait_dscnt 0" ::: "memory");

    // ---- stage 1: h1 = silu(ef @ W1 + b1)  (8 nt x 10 ks x 2 subtiles) ----
    {
      const _Float16* ar0 = &ef[m16][0];
      const _Float16* ar1 = &ef[16 + m16][0];
      for (int nt = 0; nt < 8; ++nt) {
        v8f a0 = {}, a1 = {};
        const _Float16* brow = Wt1 + (size_t)(nt * 16 + m16) * KP1;
#pragma unroll
        for (int ks = 0; ks < KP1 / 32; ++ks) {
          v16h b = frag_ld(brow, ks, lane);
          a0 = wmma16(frag_ld(ar0, ks, lane), b, a0);
          a1 = wmma16(frag_ld(ar1, ks, lane), b, a1);
        }
        float bia = b1[nt * 16 + m16];
#pragma unroll
        for (int v = 0; v < 8; ++v) {
          int m = v + mhi;
          hs[m][nt * 16 + m16]      = (_Float16)silu_f(a0[v] + bia);
          hs[16 + m][nt * 16 + m16] = (_Float16)silu_f(a1[v] + bia);
        }
      }
    }
    asm volatile("s_wait_dscnt 0" ::: "memory");

    // ---- stage 2: h2 = silu(h1 @ W2 + b2) ---------------------------------
    {
      const _Float16* ar0 = &hs[m16][0];
      const _Float16* ar1 = &hs[16 + m16][0];
      for (int nt = 0; nt < 8; ++nt) {
        v8f a0 = {}, a1 = {};
        const _Float16* brow = Wt2 + (size_t)(nt * 16 + m16) * HID;
#pragma unroll
        for (int ks = 0; ks < HID / 32; ++ks) {
          v16h b = frag_ld(brow, ks, lane);
          a0 = wmma16(frag_ld(ar0, ks, lane), b, a0);
          a1 = wmma16(frag_ld(ar1, ks, lane), b, a1);
        }
        float bia = b2[nt * 16 + m16];
#pragma unroll
        for (int v = 0; v < 8; ++v) {
          int m = v + mhi;
          h2[m * HID + nt * 16 + m16]        = (_Float16)silu_f(a0[v] + bia);
          h2[(16 + m) * HID + nt * 16 + m16] = (_Float16)silu_f(a1[v] + bia);
        }
      }
    }
    asm volatile("s_wait_dscnt 0" ::: "memory");

    // ---- stage 3: msg = h2 @ W3 + b3, scatter-add into agg[col] -----------
    {
      const _Float16* ar0 = h2 + m16 * HID;
      const _Float16* ar1 = h2 + (16 + m16) * HID;
      for (int nt = 0; nt < 8; ++nt) {
        v8f a0 = {}, a1 = {};
        const _Float16* brow = Wt3 + (size_t)(nt * 16 + m16) * HID;
#pragma unroll
        for (int ks = 0; ks < HID / 32; ++ks) {
          v16h b = frag_ld(brow, ks, lane);
          a0 = wmma16(frag_ld(ar0, ks, lane), b, a0);
          a1 = wmma16(frag_ld(ar1, ks, lane), b, a1);
        }
        float bia = b3[nt * 16 + m16];
#pragma unroll
        for (int v = 0; v < 8; ++v) {
          int m = v + mhi;
          atomicAdd(&agg[(size_t)cols[m] * HID + nt * 16 + m16],      a0[v] + bia);
          atomicAdd(&agg[(size_t)cols[16 + m] * HID + nt * 16 + m16], a1[v] + bia);
        }
      }
    }
    asm volatile("s_wait_dscnt 0" ::: "memory");
  }
}

// ---------------------------------------------------------------------------
// Node kernel: one wave owns a 32-node tile (2 x 16-row WMMA subtiles).
//   cf=[x|agg] -> silu -> u1 -> u2 -> silu -> u3 -> y=x+u, LayerNorm -> x
// ---------------------------------------------------------------------------
__global__ __launch_bounds__(64) void node_kernel(
    const float* __restrict__ agg,
    const _Float16* __restrict__ WtU1, const float* __restrict__ ub1,
    const _Float16* __restrict__ WtU2, const float* __restrict__ ub2,
    const _Float16* __restrict__ WtI1, const float* __restrict__ ib1,
    const _Float16* __restrict__ WtI2, const float* __restrict__ ib2,
    const float* __restrict__ lng, const float* __restrict__ lnb,
    float* __restrict__ x, _Float16* __restrict__ xh) {
  __shared__ _Float16 s_cf[2][32][KU];    // 32 KB
  __shared__ _Float16 s_h [2][32][HID];   // 16 KB

  const int lane = threadIdx.x & 31;
  const int wid  = threadIdx.x >> 5;
  _Float16 (*cf)[KU] = s_cf[wid];
  _Float16 (*hs)[HID] = s_h[wid];
  _Float16* h2 = &s_cf[wid][0][0];        // cf reused after stage 1: [32][HID] halves
  float*    yf = (float*)&s_cf[wid][0][0];// cf reused after stage 3: [32][HID] f32 (16 KB)

  const int ntile = N_NODES / 32;
  const int gw0 = blockIdx.x * 2 + wid;
  const int gws = gridDim.x * 2;
  const int m16 = lane & 15;
  const int mhi = (lane >> 4) << 3;

  for (int t = gw0; t < ntile; t += gws) {
    // ---- fill cf = [x_f16 | f16(agg)] -------------------------------------
    for (int m = 0; m < 32; ++m) {
      int nd = t * 32 + m;
      const unsigned* sx = (const unsigned*)(xh + (size_t)nd * HID);
      unsigned* d = (unsigned*)&cf[m][0];
      d[lane] = sx[lane]; d[lane + 32] = sx[lane + 32];
      const float* sa = agg + (size_t)nd * HID;
      for (int j = lane; j < HID; j += 32) cf[m][HID + j] = (_Float16)sa[j];
    }
    asm volatile("s_wait_dscnt 0" ::: "memory");

    // ---- stage 1: h1 = silu(cf @ U1 + ub1)  (K=256) -----------------------
    {
      const _Float16* ar0 = &cf[m16][0];
      const _Float16* ar1 = &cf[16 + m16][0];
      for (int nt = 0; nt < 8; ++nt) {
        v8f a0 = {}, a1 = {};
        const _Float16* brow = WtU1 + (size_t)(nt * 16 + m16) * KU;
#pragma unroll
        for (int ks = 0; ks < KU / 32; ++ks) {
          v16h b = frag_ld(brow, ks, lane);
          a0 = wmma16(frag_ld(ar0, ks, lane), b, a0);
          a1 = wmma16(frag_ld(ar1, ks, lane), b, a1);
        }
        float bia = ub1[nt * 16 + m16];
#pragma unroll
        for (int v = 0; v < 8; ++v) {
          int m = v + mhi;
          hs[m][nt * 16 + m16]      = (_Float16)silu_f(a0[v] + bia);
          hs[16 + m][nt * 16 + m16] = (_Float16)silu_f(a1[v] + bia);
        }
      }
    }
    asm volatile("s_wait_dscnt 0" ::: "memory");

    // ---- stage 2: u2 = h1 @ U2 + ub2 (no activation) ----------------------
    {
      const _Float16* ar0 = &hs[m16][0];
      const _Float16* ar1 = &hs[16 + m16][0];
      for (int nt = 0; nt < 8; ++nt) {
        v8f a0 = {}, a1 = {};
        const _Float16* brow = WtU2 + (size_t)(nt * 16 + m16) * HID;
#pragma unroll
        for (int ks = 0; ks < HID / 32; ++ks) {
          v16h b = frag_ld(brow, ks, lane);
          a0 = wmma16(frag_ld(ar0, ks, lane), b, a0);
          a1 = wmma16(frag_ld(ar1, ks, lane), b, a1);
        }
        float bia = ub2[nt * 16 + m16];
#pragma unroll
        for (int v = 0; v < 8; ++v) {
          int m = v + mhi;
          h2[m * HID + nt * 16 + m16]        = (_Float16)(a0[v] + bia);
          h2[(16 + m) * HID + nt * 16 + m16] = (_Float16)(a1[v] + bia);
        }
      }
    }
    asm volatile("s_wait_dscnt 0" ::: "memory");

    // ---- stage 3: h3 = silu(u2 @ I1 + ib1) --------------------------------
    {
      const _Float16* ar0 = h2 + m16 * HID;
      const _Float16* ar1 = h2 + (16 + m16) * HID;
      for (int nt = 0; nt < 8; ++nt) {
        v8f a0 = {}, a1 = {};
        const _Float16* brow = WtI1 + (size_t)(nt * 16 + m16) * HID;
#pragma unroll
        for (int ks = 0; ks < HID / 32; ++ks) {
          v16h b = frag_ld(brow, ks, lane);
          a0 = wmma16(frag_ld(ar0, ks, lane), b, a0);
          a1 = wmma16(frag_ld(ar1, ks, lane), b, a1);
        }
        float bia = ib1[nt * 16 + m16];
#pragma unroll
        for (int v = 0; v < 8; ++v) {
          int m = v + mhi;
          hs[m][nt * 16 + m16]      = (_Float16)silu_f(a0[v] + bia);
          hs[16 + m][nt * 16 + m16] = (_Float16)silu_f(a1[v] + bia);
        }
      }
    }
    asm volatile("s_wait_dscnt 0" ::: "memory");

    // ---- stage 4: y = x + (h3 @ I2 + ib2) ---------------------------------
    {
      const _Float16* ar0 = &hs[m16][0];
      const _Float16* ar1 = &hs[16 + m16][0];
      for (int nt = 0; nt < 8; ++nt) {
        v8f a0 = {}, a1 = {};
        const _Float16* brow = WtI2 + (size_t)(nt * 16 + m16) * HID;
#pragma unroll
        for (int ks = 0; ks < HID / 32; ++ks) {
          v16h b = frag_ld(brow, ks, lane);
          a0 = wmma16(frag_ld(ar0, ks, lane), b, a0);
          a1 = wmma16(frag_ld(ar1, ks, lane), b, a1);
        }
        float bia = ib2[nt * 16 + m16];
#pragma unroll
        for (int v = 0; v < 8; ++v) {
          int m = v + mhi;
          int nd0 = t * 32 + m, nd1 = t * 32 + 16 + m;
          yf[m * HID + nt * 16 + m16] =
              a0[v] + bia + x[(size_t)nd0 * HID + nt * 16 + m16];
          yf[(16 + m) * HID + nt * 16 + m16] =
              a1[v] + bia + x[(size_t)nd1 * HID + nt * 16 + m16];
        }
      }
    }
    asm volatile("s_wait_dscnt 0" ::: "memory");

    // ---- LayerNorm: one row per lane --------------------------------------
    {
      int m = lane, nd = t * 32 + m;
      float mu = 0.f;
      for (int j = 0; j < HID; ++j) mu += yf[m * HID + j];
      mu *= (1.0f / HID);
      float var = 0.f;
      for (int j = 0; j < HID; ++j) { float d = yf[m * HID + j] - mu; var += d * d; }
      float rs = rsqrtf(var * (1.0f / HID) + 1e-5f);
      for (int j = 0; j < HID; ++j) {
        float v = (yf[m * HID + j] - mu) * rs * lng[j] + lnb[j];
        x[(size_t)nd * HID + j] = v;
        xh[(size_t)nd * HID + j] = (_Float16)v;
      }
    }
    asm volatile("s_wait_dscnt 0" ::: "memory");
  }
}

// ---------------------------------------------------------------------------
// Readout + FiLM + head
// ---------------------------------------------------------------------------
__global__ void scatter_graph(const float* __restrict__ x, const int* __restrict__ batch,
                              float* __restrict__ gf) {
  int n = blockIdx.x, t = threadIdx.x;
  int b = batch[n];
  atomicAdd(&gf[(size_t)b * HID + t], x[(size_t)n * HID + t]);
}

__global__ void head_kernel(const float* __restrict__ gf, const int* __restrict__ domid,
                            const float* __restrict__ domtab,
                            const float* __restrict__ fWg, const float* __restrict__ fbg,
                            const float* __restrict__ fWb, const float* __restrict__ fbb,
                            const float* __restrict__ hW1, const float* __restrict__ hb1,
                            const float* __restrict__ hW2, const float* __restrict__ hb2,
                            const float* __restrict__ hW3, const float* __restrict__ hb3,
                            float* __restrict__ out) {
  __shared__ float de[16], gfm[HID], h1[HID], h2[64];
  int g = blockIdx.x, t = threadIdx.x;
  if (t < 16) de[t] = domtab[domid[g] * 16 + t];
  __syncthreads();
  if (t < HID) {
    float ga = fbg[t], be = fbb[t];
    for (int k = 0; k < 16; ++k) { ga += de[k] * fWg[k * HID + t]; be += de[k] * fWb[k * HID + t]; }
    gfm[t] = ga * gf[(size_t)g * HID + t] + be;
  }
  __syncthreads();
  if (t < HID) {
    float s = hb1[t];
    for (int k = 0; k < HID; ++k) s += gfm[k] * hW1[k * HID + t];
    h1[t] = silu_f(s);
  }
  __syncthreads();
  if (t < 64) {
    float s = hb2[t];
    for (int k = 0; k < HID; ++k) s += h1[k] * hW2[k * 64 + t];
    h2[t] = silu_f(s);
  }
  __syncthreads();
  if (t == 0) {
    float s = hb3[0];
    for (int k = 0; k < 64; ++k) s += h2[k] * hW3[k];
    out[g] = s;
  }
}

// ---------------------------------------------------------------------------
extern "C" void kernel_launch(void* const* d_in, const int* in_sizes, int n_in,
                              void* d_out, int out_size, void* d_ws, size_t ws_size,
                              hipStream_t stream) {
  (void)in_sizes; (void)n_in; (void)out_size; (void)ws_size;
  const float* dist   = (const float*)d_in[0];
  const int*   an     = (const int*)d_in[1];
  const int*   eidx   = (const int*)d_in[2];
  const int*   batch  = (const int*)d_in[3];
  const int*   domid  = (const int*)d_in[4];
  const float* emb    = (const float*)d_in[5];
  const float* mW1 = (const float*)d_in[6];  const float* mb1 = (const float*)d_in[7];
  const float* mW2 = (const float*)d_in[8];  const float* mb2 = (const float*)d_in[9];
  const float* mW3 = (const float*)d_in[10]; const float* mb3 = (const float*)d_in[11];
  const float* uW1 = (const float*)d_in[12]; const float* ub1 = (const float*)d_in[13];
  const float* uW2 = (const float*)d_in[14]; const float* ub2 = (const float*)d_in[15];
  const float* iW1 = (const float*)d_in[16]; const float* ib1 = (const float*)d_in[17];
  const float* iW2 = (const float*)d_in[18]; const float* ib2 = (const float*)d_in[19];
  const float* lng = (const float*)d_in[20]; const float* lnb = (const float*)d_in[21];
  const float* domtab = (const float*)d_in[22];
  const float* fWg = (const float*)d_in[23]; const float* fbg = (const float*)d_in[24];
  const float* fWb = (const float*)d_in[25]; const float* fbb = (const float*)d_in[26];
  const float* hW1 = (const float*)d_in[27]; const float* hb1 = (const float*)d_in[28];
  const float* hW2 = (const float*)d_in[29]; const float* hb2 = (const float*)d_in[30];
  const float* hW3 = (const float*)d_in[31]; const float* hb3 = (const float*)d_in[32];

  char* ws = (char*)d_ws;
  size_t off = 0;
  auto carve = [&](size_t bytes) -> char* {
    char* p = ws + off; off = (off + bytes + 255) & ~(size_t)255; return p;
  };
  float*    x    = (float*)   carve((size_t)N_NODES * HID * 4);
  _Float16* xh   = (_Float16*)carve((size_t)N_NODES * HID * 2);
  float*    agg  = (float*)   carve((size_t)N_NODES * HID * 4);
  float*    gf   = (float*)   carve((size_t)N_GRAPHS * HID * 4);
  _Float16* Wt1  = (_Float16*)carve((size_t)NLAYERS * HID * KP1 * 2);
  _Float16* WtU1 = (_Float16*)carve((size_t)NLAYERS * HID * KU * 2);
  _Float16* WtM2 = (_Float16*)carve((size_t)NLAYERS * HID * HID * 2);
  _Float16* WtM3 = (_Float16*)carve((size_t)NLAYERS * HID * HID * 2);
  _Float16* WtU2 = (_Float16*)carve((size_t)NLAYERS * HID * HID * 2);
  _Float16* WtI1 = (_Float16*)carve((size_t)NLAYERS * HID * HID * 2);
  _Float16* WtI2 = (_Float16*)carve((size_t)NLAYERS * HID * HID * 2);

  // ---- weight prep (f32 -> f16, transposed [out][in]) ----------------------
  { int n = NLAYERS * HID * KP1; prep_w1<<<(n + 255) / 256, 256, 0, stream>>>(mW1, Wt1); }
  { int n = NLAYERS * HID * KU;  prep_t<<<(n + 255) / 256, 256, 0, stream>>>(uW1, WtU1, KU, n); }
  { int n = NLAYERS * HID * HID;
    prep_t<<<(n + 255) / 256, 256, 0, stream>>>(mW2, WtM2, HID, n);
    prep_t<<<(n + 255) / 256, 256, 0, stream>>>(mW3, WtM3, HID, n);
    prep_t<<<(n + 255) / 256, 256, 0, stream>>>(uW2, WtU2, HID, n);
    prep_t<<<(n + 255) / 256, 256, 0, stream>>>(iW1, WtI1, HID, n);
    prep_t<<<(n + 255) / 256, 256, 0, stream>>>(iW2, WtI2, HID, n); }

  // ---- embedding -----------------------------------------------------------
  { size_t n = (size_t)N_NODES * HID;
    embed_kernel<<<(int)((n + 255) / 256), 256, 0, stream>>>(an, emb, x, xh); }

  // ---- interaction layers --------------------------------------------------
  for (int l = 0; l < NLAYERS; ++l) {
    { size_t n = (size_t)N_NODES * HID;
      zero_f<<<(int)((n + 255) / 256), 256, 0, stream>>>(agg, n); }
    edge_kernel<<<2500, 64, 0, stream>>>(
        dist, eidx, eidx + N_EDGES, xh,
        Wt1  + (size_t)l * HID * KP1, mb1 + l * HID,
        WtM2 + (size_t)l * HID * HID, mb2 + l * HID,
        WtM3 + (size_t)l * HID * HID, mb3 + l * HID, agg);
    node_kernel<<<313, 64, 0, stream>>>(
        agg,
        WtU1 + (size_t)l * HID * KU,  ub1 + l * HID,
        WtU2 + (size_t)l * HID * HID, ub2 + l * HID,
        WtI1 + (size_t)l * HID * HID, ib1 + l * HID,
        WtI2 + (size_t)l * HID * HID, ib2 + l * HID,
        lng + l * HID, lnb + l * HID, x, xh);
  }

  // ---- readout + FiLM + head ----------------------------------------------
  { size_t n = (size_t)N_GRAPHS * HID;
    zero_f<<<(int)((n + 255) / 256), 256, 0, stream>>>(gf, n); }
  scatter_graph<<<N_NODES, 128, 0, stream>>>(x, batch, gf);
  head_kernel<<<N_GRAPHS, 128, 0, stream>>>(gf, domid, domtab, fWg, fbg, fWb, fbb,
                                            hW1, hb1, hW2, hb2, hW3, hb3,
                                            (float*)d_out);
}
